// HST_292057776452
// MI455X (gfx1250) — matrix operand order repaired and checked by reading, not compiled, not run on hardware
//
#include <hip/hip_runtime.h>
#include <cstdint>
#include <cstddef>

// ---------------------------------------------------------------------------
// MI455X (gfx1250) implementation of the 2-block hierarchical window
// transformer. All five large GEMMs run on v_wmma_f32_16x16x32_f16 with the
// A-tile staged into LDS by the Tensor Data Mover (tensor_load_to_lds).
// Activations f16, accumulation f32, residual stream f32.
// ---------------------------------------------------------------------------

typedef unsigned short ushort_t;
typedef unsigned int   uint_t;

typedef _Float16 f16_t;
typedef f16_t v16h __attribute__((ext_vector_type(16)));
typedef float v8f  __attribute__((ext_vector_type(8)));
typedef unsigned int u32x4 __attribute__((ext_vector_type(4)));
typedef int          i32x8 __attribute__((ext_vector_type(8)));
typedef int          i32x4 __attribute__((ext_vector_type(4)));

union Frag {
  uint4 q[2];   // 8 dwords = 16 f16
  v16h  v;
};

// ----- constants ------------------------------------------------------------
#define BB   32
#define CC   512
#define HH   32
#define WW   32
#define NTOK (BB*HH*WW)          // 32768
#define H2   4
#define DH   64
#define HALF 256
#define SL   64
#define NWIN 16

// GEMM tiling
#define KC       512              // K-chunk staged in LDS per TDM op
#define LDS_ROWW 260              // dwords per LDS A row (256 + 4 pad = 1040B)

__device__ __forceinline__ float h2f(ushort_t u) {
  union { ushort_t u; f16_t h; } c; c.u = u; return (float)c.h;
}
__device__ __forceinline__ ushort_t f2h(float f) {
  union { f16_t h; ushort_t u; } c; c.h = (f16_t)f; return c.u;
}
__device__ __forceinline__ float gelu_exact(float x) {
  return 0.5f * x * (1.0f + erff(x * 0.70710678118654752440f));
}
__device__ __forceinline__ int tok_row(int b, int win, int sl) {
  int h = (win >> 2) * 8 + (sl >> 3);
  int w = (win & 3) * 8 + (sl & 7);
  return (b * HH + h) * WW + w;
}

// ----- Tensor Data Mover: 2-D tile (64 rows x 1024B) global -> LDS ----------
// D# per CDNA5 ISA ch.8. data_size=8B. Pad 4 dwords after every 256 dwords so
// LDS rows are skewed 16B (bank-conflict-free ds_load_b128 fragment reads).
// This toolchain's builtin takes 6 args: (g0 v4u, g1 v8i, g2 v4i, g3 v4i,
// extra v8i, cpol i32) -- unused trailing groups are zero-filled.
__device__ __forceinline__ void tdm_load_tile(const void* gaddr, uint_t lds_off,
                                              int row_stride_8b) {
  unsigned long long ga = (unsigned long long)(size_t)gaddr;
  u32x4 g0;
  g0[0] = 1u;                                        // count=1, user mode
  g0[1] = lds_off;                                   // lds_addr (bytes)
  g0[2] = (uint_t)(ga & 0xFFFFFFFFu);                // global_addr[31:0]
  g0[3] = (uint_t)((ga >> 32) & 0x01FFFFFFu)         // global_addr[56:32]
        | (2u << 30);                                // type = 2 ("image")
  i32x8 g1;
  g1[0] = (int)((3u << 16)                           // data_size = 8B
              | (1u << 20)                           // pad_enable
              | (7u << 22)                           // pad_interval: 256 dw
              | (3u << 25));                         // pad_amount: 4 dw
  g1[1] = 0;                                         // tdim0 low16 (2^30 -> 0)
  g1[2] = 0x4000;                                    // tdim0 hi / tdim1 lo
  g1[3] = (int)(0x4000u | (128u << 16));             // tdim1 hi | tile_dim0=128
  g1[4] = 64;                                        // tile_dim1=64, tile_dim2=0
  g1[5] = row_stride_8b;                             // tensor_dim0_stride lo32
  g1[6] = 0;
  g1[7] = 0;
  i32x4 gz4 = {0, 0, 0, 0};
  i32x8 gz8 = {0, 0, 0, 0, 0, 0, 0, 0};
  __builtin_amdgcn_tensor_load_to_lds(g0, g1, gz4, gz4, gz8, 0);
}

// ----- layout transposes ----------------------------------------------------
__global__ void k_nchw_to_nhwc(const float* __restrict__ in, float* __restrict__ out) {
  size_t i = (size_t)blockIdx.x * blockDim.x + threadIdx.x;     // out-linear
  int c = i & (CC - 1); size_t t = i >> 9;
  int w = t & (WW - 1); t >>= 5; int h = t & (HH - 1); int b = (int)(t >> 5);
  out[i] = in[(((size_t)b * CC + c) * HH + h) * WW + w];
}
__global__ void k_nhwc_to_nchw(const float* __restrict__ in, float* __restrict__ out) {
  size_t i = (size_t)blockIdx.x * blockDim.x + threadIdx.x;     // out-linear
  int w = i & (WW - 1); size_t t = i >> 5;
  int h = t & (HH - 1); t >>= 5; int c = t & (CC - 1); int b = (int)(t >> 9);
  out[i] = in[(((size_t)b * HH + h) * WW + w) * CC + c];
}

// ----- LayerNorm: one wave32 per row of 512 ---------------------------------
__global__ __launch_bounds__(256) void k_layernorm(
    const float* __restrict__ Y, const float* __restrict__ g,
    const float* __restrict__ bta, ushort_t* __restrict__ out) {
  int lane = threadIdx.x & 31;
  int wave = threadIdx.x >> 5;
  int row  = blockIdx.x * 8 + wave;
  const float4* s4 = (const float4*)(Y + (size_t)row * CC + lane * 16);
  float4 f0 = s4[0], f1 = s4[1], f2 = s4[2], f3 = s4[3];
  float v[16] = {f0.x,f0.y,f0.z,f0.w, f1.x,f1.y,f1.z,f1.w,
                 f2.x,f2.y,f2.z,f2.w, f3.x,f3.y,f3.z,f3.w};
  float s = 0.f, q = 0.f;
  #pragma unroll
  for (int e = 0; e < 16; ++e) { s += v[e]; q += v[e]*v[e]; }
  #pragma unroll
  for (int off = 16; off > 0; off >>= 1) {
    s += __shfl_xor(s, off, 32);
    q += __shfl_xor(q, off, 32);
  }
  float m   = s * (1.0f / CC);
  float var = q * (1.0f / CC) - m * m;
  float inv = rsqrtf(var + 1e-5f);
  uint_t* o = (uint_t*)(out + (size_t)row * CC) + lane * 8;
  #pragma unroll
  for (int p = 0; p < 8; ++p) {
    int c0 = lane * 16 + 2 * p;
    float a = (v[2*p]   - m) * inv * g[c0]   + bta[c0];
    float b = (v[2*p+1] - m) * inv * g[c0+1] + bta[c0+1];
    o[p] = ((uint_t)f2h(b) << 16) | (uint_t)f2h(a);
  }
}

// ----- pack fp32 weights into WMMA B-fragment order -------------------------
// dst index: (((kb*(Ntot/16) + nbGlobal)*32 + lane)*8 + p) dwords.
// lane: n = lane&15, hi = lane>>4 ; element pair p covers K = kb*32 + hi*16 + 2p.
__global__ void k_pack(const float* __restrict__ src, uint_t* __restrict__ dst,
                       int K, int Nsrc, int Ntot, int ncolOff, int total) {
  int idx = blockIdx.x * blockDim.x + threadIdx.x;
  if (idx >= total) return;
  int p    = idx & 7;
  int lane = (idx >> 3) & 31;
  int rest = idx >> 8;
  int nbN  = Nsrc >> 4;
  int nbs  = rest % nbN;
  int kb   = rest / nbN;
  int hi   = lane >> 4, ln = lane & 15;
  int K0   = kb * 32 + hi * 16 + 2 * p;
  int n    = nbs * 16 + ln;
  ushort_t lo  = f2h(src[(size_t)K0       * Nsrc + n]);
  ushort_t hiv = f2h(src[(size_t)(K0 + 1) * Nsrc + n]);
  int nbg = (ncolOff >> 4) + nbs;
  dst[(((size_t)kb * (Ntot >> 4) + nbg) * 32 + lane) * 8 + p] =
      ((uint_t)hiv << 16) | (uint_t)lo;
}

// ----- WMMA GEMM: C[M,Nt] = A[M,K] (f16) @ Wpacked ; fused epilogue ---------
// block = 256 threads = 8 waves (2 M x 4 N). Wave tile 32 x 64 (2x4 frags).
// Block tile 64(M) x 256(N). A chunk (64 x 512 f16) staged in LDS by the TDM.
__global__ __launch_bounds__(256) void k_gemm_wmma(
    const uint_t* __restrict__ A,      // f16 pairs, [M, K/2] dwords
    const uint_t* __restrict__ W,      // packed fragments
    const float*  __restrict__ bias,   // or nullptr
    float*        __restrict__ resid,  // fp32 [M,Nt] accumulate, or nullptr
    ushort_t*     __restrict__ obf,    // f16 [M,Nt] output, or nullptr
    int K, int Nt, int fuse_gelu) {
  __shared__ uint_t sA[64 * LDS_ROWW];               // 65 KB, 16B-skewed rows

  const int lane  = threadIdx.x & 31;
  const int wave  = threadIdx.x >> 5;
  const int wm    = wave >> 2, wn = wave & 3;
  const int mbase = blockIdx.y * 64 + wm * 32;
  const int nbase = blockIdx.x * 256 + wn * 64;
  const int lm    = lane & 15, hi = lane >> 4;
  const int nb16  = Nt >> 4;
  const int nblk0 = nbase >> 4;

  const uint_t* a0 = sA + (size_t)(wm * 32 + lm)      * LDS_ROWW;
  const uint_t* a1 = sA + (size_t)(wm * 32 + 16 + lm) * LDS_ROWW;

  v8f acc[2][4];
  #pragma unroll
  for (int mf = 0; mf < 2; ++mf)
    #pragma unroll
    for (int nb = 0; nb < 4; ++nb)
      acc[mf][nb] = (v8f){0.f,0.f,0.f,0.f,0.f,0.f,0.f,0.f};

  const int nchunk = K / KC;
  for (int ch = 0; ch < nchunk; ++ch) {
    if (wave == 0) {                                  // one TDM op per block
      const uint_t* ga = A + (size_t)(blockIdx.y * 64) * (K >> 1) + (size_t)ch * (KC >> 1);
      tdm_load_tile(ga, (uint_t)(size_t)sA, K >> 2);  // row stride in 8B units
      __builtin_amdgcn_s_wait_tensorcnt(0);
    }
    __syncthreads();

    #pragma unroll 2
    for (int kb = 0; kb < (KC >> 5); ++kb) {
      Frag af0, af1;
      af0.q[0] = *(const uint4*)(a0 + kb * 16 + hi * 8);
      af0.q[1] = *(const uint4*)(a0 + kb * 16 + hi * 8 + 4);
      af1.q[0] = *(const uint4*)(a1 + kb * 16 + hi * 8);
      af1.q[1] = *(const uint4*)(a1 + kb * 16 + hi * 8 + 4);
      const int kbg = ch * (KC >> 5) + kb;
      const uint_t* wb = W + (((size_t)kbg * nb16 + nblk0) * 32 + lane) * 8;
      __builtin_prefetch(wb + (size_t)nb16 * 256, 0, 3);   // next 32-k step of B
      #pragma unroll
      for (int nb = 0; nb < 4; ++nb) {
        Frag bf;
        const uint4* wp = (const uint4*)(wb + nb * 256);
        bf.q[0] = wp[0];
        bf.q[1] = wp[1];
        acc[0][nb] = __builtin_amdgcn_wmma_f32_16x16x32_f16(
            false, af0.v, false, bf.v, (short)0, acc[0][nb], false, false);
        acc[1][nb] = __builtin_amdgcn_wmma_f32_16x16x32_f16(
            false, af1.v, false, bf.v, (short)0, acc[1][nb], false, false);
      }
    }
    __syncthreads();                                  // before LDS reuse
  }

  // Epilogue. C/D layout: lane (n=lm, hi) ; vgpr r -> row += hi*8 + r.
  #pragma unroll
  for (int mf = 0; mf < 2; ++mf) {
    #pragma unroll
    for (int nb = 0; nb < 4; ++nb) {
      int col = nbase + nb * 16 + lm;
      #pragma unroll
      for (int r = 0; r < 8; ++r) {
        int row = mbase + mf * 16 + hi * 8 + r;
        float val = acc[mf][nb][r];
        if (fuse_gelu) val = gelu_exact(val);
        if (bias)      val += bias[col];
        size_t idx = (size_t)row * Nt + col;
        if (resid) resid[idx] += val;
        if (obf)   obf[idx] = f2h(val);
      }
    }
  }
}

// ----- local window attention: block = (b,win), blockIdx.y = head -----------
__global__ __launch_bounds__(64) void k_attn_local(
    const ushort_t* __restrict__ qkv, const float* __restrict__ pos1,
    ushort_t* __restrict__ out) {
  __shared__ ushort_t ks[SL][DH];
  __shared__ ushort_t vs[SL][DH];
  __shared__ float    scs[SL][SL + 1];
  int bw = blockIdx.x, head = blockIdx.y;
  int b = bw >> 4, win = bw & 15;
  int i = threadIdx.x;                    // query index 0..63
  int r_i = tok_row(b, win, i);
  {
    const uint_t* kp = (const uint_t*)(qkv + (size_t)r_i * 1536 + 512  + head * DH);
    const uint_t* vp = (const uint_t*)(qkv + (size_t)r_i * 1536 + 1024 + head * DH);
    uint_t* kd = (uint_t*)ks[i];
    uint_t* vd = (uint_t*)vs[i];
    #pragma unroll 8
    for (int t = 0; t < 32; ++t) { kd[t] = kp[t]; vd[t] = vp[t]; }
  }
  __syncthreads();
  const ushort_t* qp = qkv + (size_t)r_i * 1536 + head * DH;
  float qf[DH];
  #pragma unroll 8
  for (int d = 0; d < DH; ++d) qf[d] = h2f(qp[d]) * 0.125f;   // SCALE = 64^-0.5
  float mx = -1e30f;
  for (int j = 0; j < SL; ++j) {
    float s = 0.f;
    #pragma unroll 8
    for (int d = 0; d < DH; ++d) s += qf[d] * h2f(ks[j][d]);
    s += pos1[((size_t)head * SL + i) * SL + j];
    scs[i][j] = s;
    mx = fmaxf(mx, s);
  }
  float l = 0.f;
  for (int j = 0; j < SL; ++j) {
    float e = __expf(scs[i][j] - mx);
    scs[i][j] = e;
    l += e;
  }
  float o[DH];
  #pragma unroll 8
  for (int d = 0; d < DH; ++d) o[d] = 0.f;
  for (int j = 0; j < SL; ++j) {
    float p = scs[i][j];
    #pragma unroll 8
    for (int d = 0; d < DH; ++d) o[d] += p * h2f(vs[j][d]);
  }
  float inv = 1.0f / l;
  ushort_t* op = out + (size_t)r_i * CC + head * DH;
  #pragma unroll 8
  for (int d = 0; d < DH; ++d) op[d] = f2h(o[d] * inv);
}

// ----- global (window-level) attention: block = (b,sl), blockIdx.y = head ---
__global__ __launch_bounds__(32) void k_attn_global(
    const ushort_t* __restrict__ qkv, const float* __restrict__ pos2,
    ushort_t* __restrict__ out) {
  __shared__ ushort_t ks[NWIN][DH];
  __shared__ ushort_t vs[NWIN][DH];
  int bx = blockIdx.x, head = blockIdx.y;
  int b = bx >> 6, sl = bx & 63;
  int i = threadIdx.x;                    // window index 0..15 (active)
  if (i < NWIN) {
    int r = tok_row(b, i, sl);
    const uint_t* kp = (const uint_t*)(qkv + (size_t)r * 1536 + 768  + head * DH);
    const uint_t* vp = (const uint_t*)(qkv + (size_t)r * 1536 + 1280 + head * DH);
    uint_t* kd = (uint_t*)ks[i];
    uint_t* vd = (uint_t*)vs[i];
    #pragma unroll 8
    for (int t = 0; t < 32; ++t) { kd[t] = kp[t]; vd[t] = vp[t]; }
  }
  __syncthreads();
  if (i >= NWIN) return;
  int r_i = tok_row(b, i, sl);
  const ushort_t* qp = qkv + (size_t)r_i * 1536 + HALF + head * DH;
  float qf[DH];
  #pragma unroll 8
  for (int d = 0; d < DH; ++d) qf[d] = h2f(qp[d]) * 0.125f;
  float sc[NWIN];
  float mx = -1e30f;
  for (int j = 0; j < NWIN; ++j) {
    float s = 0.f;
    #pragma unroll 8
    for (int d = 0; d < DH; ++d) s += qf[d] * h2f(ks[j][d]);
    s += pos2[((size_t)head * NWIN + i) * NWIN + j];
    sc[j] = s;
    mx = fmaxf(mx, s);
  }
  float l = 0.f;
  #pragma unroll
  for (int j = 0; j < NWIN; ++j) { sc[j] = __expf(sc[j] - mx); l += sc[j]; }
  float o[DH];
  #pragma unroll 8
  for (int d = 0; d < DH; ++d) o[d] = 0.f;
  for (int j = 0; j < NWIN; ++j) {
    float p = sc[j];
    #pragma unroll 8
    for (int d = 0; d < DH; ++d) o[d] += p * h2f(vs[j][d]);
  }
  float inv = 1.0f / l;
  ushort_t* op = out + (size_t)r_i * CC + HALF + head * DH;
  #pragma unroll 8
  for (int d = 0; d < DH; ++d) op[d] = f2h(o[d] * inv);
}

// ----- 3x3 depthwise conv (SAME, groups = 2048) + GELU ----------------------
__global__ void k_dwconv(const ushort_t* __restrict__ y1,
                         const float* __restrict__ wd,
                         ushort_t* __restrict__ y2) {
  size_t idx = (size_t)blockIdx.x * blockDim.x + threadIdx.x;  // B*H*W*2048
  int c = idx & 2047; size_t t = idx >> 11;
  int w = t & (WW - 1); t >>= 5; int h = t & (HH - 1); int b = (int)(t >> 5);
  float acc = 0.f;
  #pragma unroll
  for (int ky = 0; ky < 3; ++ky) {
    int hh = h + ky - 1;
    if ((unsigned)hh >= HH) continue;
    #pragma unroll
    for (int kx = 0; kx < 3; ++kx) {
      int ww = w + kx - 1;
      if ((unsigned)ww >= WW) continue;
      acc += h2f(y1[(((size_t)(b * HH + hh)) * WW + ww) * 2048 + c]) *
             wd[(ky * 3 + kx) * 2048 + c];
    }
  }
  y2[idx] = f2h(gelu_exact(acc));
}

// ---------------------------------------------------------------------------
extern "C" void kernel_launch(void* const* d_in, const int* in_sizes, int n_in,
                              void* d_out, int out_size, void* d_ws, size_t ws_size,
                              hipStream_t stream) {
  (void)in_sizes; (void)n_in; (void)out_size; (void)ws_size;
  const float* x = (const float*)d_in[0];

  // workspace carve (~452 MB)
  char* base = (char*)d_ws;
  float*    Y   = (float*)base;    base += (size_t)NTOK * CC   * 4;   // 64 MB fp32 residual (NHWC)
  ushort_t* act = (ushort_t*)base; base += (size_t)NTOK * CC   * 2;   // 32 MB f16: ln1/attn_out/ln2
  ushort_t* qkv = (ushort_t*)base; base += (size_t)NTOK * 1536 * 2;   // 96 MB f16 q|k|v
  ushort_t* y1  = (ushort_t*)base; base += (size_t)NTOK * 2048 * 2;   // 128 MB
  ushort_t* y2  = (ushort_t*)base; base += (size_t)NTOK * 2048 * 2;   // 128 MB
  uint_t*   PW  = (uint_t*)base;                                      // 2 MB packed weights (reused)

  k_nchw_to_nhwc<<<65536, 256, 0, stream>>>(x, Y);

  for (int blk = 0; blk < 2; ++blk) {
    int ib = 1 + blk * 13;
    const float* ln1_g = (const float*)d_in[ib + 0];
    const float* ln1_b = (const float*)d_in[ib + 1];
    const float* wq    = (const float*)d_in[ib + 2];
    const float* wkv   = (const float*)d_in[ib + 3];
    const float* wo    = (const float*)d_in[ib + 4];
    const float* bo    = (const float*)d_in[ib + 5];
    const float* pos1  = (const float*)d_in[ib + 6];
    const float* pos2  = (const float*)d_in[ib + 7];
    const float* ln2_g = (const float*)d_in[ib + 8];
    const float* ln2_b = (const float*)d_in[ib + 9];
    const float* w1    = (const float*)d_in[ib + 10];
    const float* wd    = (const float*)d_in[ib + 11];
    const float* w2    = (const float*)d_in[ib + 12];

    // --- attention half ---
    k_layernorm<<<NTOK / 8, 256, 0, stream>>>(Y, ln1_g, ln1_b, act);

    int tq = 16 * 32 * 256;           // (512/32)*(512/16)*32*8
    k_pack<<<(tq + 255) / 256, 256, 0, stream>>>(wq,  PW, 512, 512,  1536, 0,   tq);
    int tk = 16 * 64 * 256;
    k_pack<<<(tk + 255) / 256, 256, 0, stream>>>(wkv, PW, 512, 1024, 1536, 512, tk);
    k_gemm_wmma<<<dim3(6, 512), 256, 0, stream>>>(
        (const uint_t*)act, PW, nullptr, nullptr, qkv, 512, 1536, 0);

    k_attn_local <<<dim3(BB * NWIN, H2), 64, 0, stream>>>(qkv, pos1, act);
    k_attn_global<<<dim3(BB * SL,  H2), 32, 0, stream>>>(qkv, pos2, act);

    int to = 16 * 32 * 256;
    k_pack<<<(to + 255) / 256, 256, 0, stream>>>(wo, PW, 512, 512, 512, 0, to);
    k_gemm_wmma<<<dim3(2, 512), 256, 0, stream>>>(
        (const uint_t*)act, PW, bo, Y, nullptr, 512, 512, 0);  // Y += attn@wo + bo

    // --- FFN half ---
    k_layernorm<<<NTOK / 8, 256, 0, stream>>>(Y, ln2_g, ln2_b, act);

    int t1 = 16 * 128 * 256;
    k_pack<<<(t1 + 255) / 256, 256, 0, stream>>>(w1, PW, 512, 2048, 2048, 0, t1);
    k_gemm_wmma<<<dim3(8, 512), 256, 0, stream>>>(
        (const uint_t*)act, PW, nullptr, nullptr, y1, 512, 2048, 1);  // GELU fused

    k_dwconv<<<(NTOK * 2048) / 256, 256, 0, stream>>>(y1, wd, y2);

    int t2 = 64 * 32 * 256;
    k_pack<<<(t2 + 255) / 256, 256, 0, stream>>>(w2, PW, 2048, 512, 512, 0, t2);
    k_gemm_wmma<<<dim3(2, 512), 256, 0, stream>>>(
        (const uint_t*)y2, PW, nullptr, Y, nullptr, 2048, 512, 0);    // Y += ffn
  }

  k_nhwc_to_nchw<<<65536, 256, 0, stream>>>(Y, (float*)d_out);
}
